// MoELinear_10282151706765
// MI455X (gfx1250) — matrix operand order repaired and checked by reading, not compiled, not run on hardware
//
#include <hip/hip_runtime.h>

typedef __attribute__((ext_vector_type(16))) __bf16 v16bf;
typedef __attribute__((ext_vector_type(8)))  __bf16 v8bf;
typedef __attribute__((ext_vector_type(4)))  __bf16 v4bf;
typedef __attribute__((ext_vector_type(8)))  float  v8f;

#define TILE_M 128
#define TILE_N 256
#define TILE_K 32
#define LP     40   // LDS row pitch in bf16 elems (80B) -> conflict-free b128 reads

// ---------------------------------------------------------------------------
// fp32 -> bf16 converter (vectorized 4-wide)
// ---------------------------------------------------------------------------
__global__ void cvt_bf16_kernel(const float4* __restrict__ in,
                                v4bf* __restrict__ out, int n4) {
    int i = blockIdx.x * blockDim.x + threadIdx.x;
    if (i < n4) {
        float4 f = in[i];
        v4bf o;
        o[0] = (__bf16)f.x; o[1] = (__bf16)f.y;
        o[2] = (__bf16)f.z; o[3] = (__bf16)f.w;
        out[i] = o;
    }
}

// ---------------------------------------------------------------------------
// Gate: one wave (32 lanes) per token; lane e owns expert e.
// logits -> softmax -> top-2 -> renormalize -> fold in SCALING.
// ---------------------------------------------------------------------------
__global__ __launch_bounds__(256) void gate_kernel(
    const float* __restrict__ x, const float* __restrict__ gw,
    float* __restrict__ wscale, int ntok, int D, float scaling) {
    int wave = (blockIdx.x * blockDim.x + threadIdx.x) >> 5;
    int lane = threadIdx.x & 31;
    if (wave >= ntok) return;
    const float* xr = x + (size_t)wave * D;
    const float* gr = gw + (size_t)lane * D;
    float acc = 0.f;
    for (int d = 0; d < D; d += 4) {
        float4 xv = *(const float4*)(xr + d);
        float4 gv = *(const float4*)(gr + d);
        acc += xv.x * gv.x + xv.y * gv.y + xv.z * gv.z + xv.w * gv.w;
    }
    // softmax over 32 lanes
    float m = acc;
    for (int o = 16; o; o >>= 1) m = fmaxf(m, __shfl_xor(m, o, 32));
    float p = __expf(acc - m);
    float s = p;
    for (int o = 16; o; o >>= 1) s += __shfl_xor(s, o, 32);
    float prob = p / s;
    // argmax #1
    float v = prob; int idx = lane;
    for (int o = 16; o; o >>= 1) {
        float ov = __shfl_xor(v, o, 32); int oi = __shfl_xor(idx, o, 32);
        if (ov > v || (ov == v && oi < idx)) { v = ov; idx = oi; }
    }
    int i1 = idx; float v1 = v;
    // argmax #2
    v = (lane == i1) ? -1.f : prob; idx = lane;
    for (int o = 16; o; o >>= 1) {
        float ov = __shfl_xor(v, o, 32); int oi = __shfl_xor(idx, o, 32);
        if (ov > v || (ov == v && oi < idx)) { v = ov; idx = oi; }
    }
    int i2 = idx; float v2 = v;
    float wnorm = scaling / (v1 + v2);
    float w = (lane == i1) ? v1 * wnorm : (lane == i2) ? v2 * wnorm : 0.f;
    wscale[(size_t)wave * 32 + lane] = w;
}

// ---------------------------------------------------------------------------
// Async ROWSx32 bf16 tile load: global -> LDS (global_load_async_to_lds_b128)
// ROWS*4 16B chunks over 256 threads. Tracked by ASYNCcnt.
// ---------------------------------------------------------------------------
template <int ROWS>
__device__ __forceinline__ void async_tile_load(
    const __bf16* __restrict__ gsrc, int ld, int rowBase, int kBase,
    __bf16* sdst, int tid) {
#pragma unroll
    for (int t = 0; t < (ROWS * 4) / 256; ++t) {
        int c   = tid + t * 256;
        int row = c >> 2;
        int cc  = c & 3;
        const __bf16* g = gsrc + (size_t)(rowBase + row) * ld + kBase + cc * 8;
        unsigned l = (unsigned)(unsigned long long)(sdst + row * LP + cc * 8);
        asm volatile("global_load_async_to_lds_b128 %0, %1, off"
                     :: "v"(l), "v"(g) : "memory");
    }
}

union ABfrag { v16bf v; v8bf h[2]; };

// Per-wave: 4(M) x 4(N) 16x16 WMMA subtiles for one K=32 step.
__device__ __forceinline__ void wave_compute(
    const __bf16* As, const __bf16* Bs, int mOff, int nOff, int lane,
    v8f acc[4][4]) {
    int r16 = lane & 15;
    int hi  = lane >> 4;
    ABfrag a[4], b[4];
#pragma unroll
    for (int i = 0; i < 4; ++i) {
        // A 16x32 bf16 layout: lane m=r16; lo lanes K=0..7/16..23, hi lanes +8
        const __bf16* p = As + (mOff + i * 16 + r16) * LP;
        a[i].h[0] = *(const v8bf*)(p + hi * 8);
        a[i].h[1] = *(const v8bf*)(p + 16 + hi * 8);
    }
#pragma unroll
    for (int j = 0; j < 4; ++j) {
        // B 32x16 bf16 layout: lane n=r16; lo lanes K=0..7/16..23, hi lanes +8
        const __bf16* p = Bs + (nOff + j * 16 + r16) * LP;
        b[j].h[0] = *(const v8bf*)(p + hi * 8);
        b[j].h[1] = *(const v8bf*)(p + 16 + hi * 8);
    }
#pragma unroll
    for (int i = 0; i < 4; ++i)
#pragma unroll
        for (int j = 0; j < 4; ++j)
            acc[i][j] = __builtin_amdgcn_wmma_f32_16x16x32_bf16(
                false, a[i].v, false, b[j].v, (short)0, acc[i][j], false, false);
}

// ---------------------------------------------------------------------------
// Two-segment fused GEMM: out[m,n] = sum_k A0[m,k]B0[n,k] + sum_k A1[m,k]B1[n,k]
// 128x256 tile / WG, 8 waves (2 in M x 4 in N), each wave 64x64 (16 WMMA/step),
// double-buffered async LDS staging.
// Epilogue: if scl != nullptr, writes bf16 acc*scl[m*32 + (n>>4)] (ax pass);
// else writes fp32 (final output).
// ---------------------------------------------------------------------------
__global__ __launch_bounds__(256, 1) void wmma_gemm_kernel(
    const __bf16* __restrict__ A0, const __bf16* __restrict__ B0, int K0,
    const __bf16* __restrict__ A1, const __bf16* __restrict__ B1, int K1,
    void* __restrict__ out_p, int N, const float* __restrict__ scl) {
    // [buf][ A: TILE_M rows | B: TILE_N rows ]
    __shared__ __bf16 smem[2][(TILE_M + TILE_N) * LP];

    int tid   = threadIdx.x;
    int lane  = tid & 31;
    int wave  = tid >> 5;
    int mBase = blockIdx.y * TILE_M;
    int nBase = blockIdx.x * TILE_N;
    int mOff  = (wave >> 2) * 64;  // 2 waves along M
    int nOff  = (wave & 3) * 64;   // 4 waves along N

    int T0 = K0 >> 5;
    int T1 = K1 >> 5;
    int T  = T0 + T1;

    v8f acc[4][4];
#pragma unroll
    for (int i = 0; i < 4; ++i)
#pragma unroll
        for (int j = 0; j < 4; ++j)
            acc[i][j] = (v8f){0.f, 0.f, 0.f, 0.f, 0.f, 0.f, 0.f, 0.f};

    auto issue = [&](int buf, int t) {
        const __bf16 *A, *B;
        int ld, kb;
        if (t < T0) { A = A0; B = B0; ld = K0; kb = t * 32; }
        else        { A = A1; B = B1; ld = K1; kb = (t - T0) * 32; }
        async_tile_load<TILE_M>(A, ld, mBase, kb, &smem[buf][0], tid);
        async_tile_load<TILE_N>(B, ld, nBase, kb, &smem[buf][TILE_M * LP], tid);
    };

    issue(0, 0);
    int cur = 0;
    for (int t = 0; t < T; ++t) {
        if (t + 1 < T) {
            issue(cur ^ 1, t + 1);
            // 6 new ops in flight; <=6 means the 6 for buffer `cur` completed
            asm volatile("s_wait_asynccnt 0x6" ::: "memory");
        } else {
            asm volatile("s_wait_asynccnt 0x0" ::: "memory");
        }
        __syncthreads();
        wave_compute(&smem[cur][0], &smem[cur][TILE_M * LP], mOff, nOff, lane, acc);
        __syncthreads();
        cur ^= 1;
    }

    // Epilogue: C/D layout — VGPR v, lanes 0-15 -> M=v, lanes 16-31 -> M=8+v
    if (scl != nullptr) {
        __bf16* outb = (__bf16*)out_p;
#pragma unroll
        for (int i = 0; i < 4; ++i)
#pragma unroll
            for (int j = 0; j < 4; ++j) {
                int n    = nBase + nOff + j * 16 + (lane & 15);
                int mrow = mBase + mOff + i * 16 + (lane >> 4) * 8;
#pragma unroll
                for (int v = 0; v < 8; ++v) {
                    float s = scl[(size_t)(mrow + v) * 32 + (n >> 4)];
                    outb[(size_t)(mrow + v) * N + n] = (__bf16)(acc[i][j][v] * s);
                }
            }
    } else {
        float* outf = (float*)out_p;
#pragma unroll
        for (int i = 0; i < 4; ++i)
#pragma unroll
            for (int j = 0; j < 4; ++j) {
                int n    = nBase + nOff + j * 16 + (lane & 15);
                int mrow = mBase + mOff + i * 16 + (lane >> 4) * 8;
#pragma unroll
                for (int v = 0; v < 8; ++v)
                    outf[(size_t)(mrow + v) * N + n] = acc[i][j][v];
            }
    }
}

// ---------------------------------------------------------------------------
// Host launch
// ---------------------------------------------------------------------------
extern "C" void kernel_launch(void* const* d_in, const int* in_sizes, int n_in,
                              void* d_out, int out_size, void* d_ws, size_t ws_size,
                              hipStream_t stream) {
    constexpr int Bv = 4, S = 2048, DIN = 4096, DOUT = 4096, E = 32, R = 16;
    constexpr int M = Bv * S;           // 8192 tokens
    constexpr int RM = E * R;           // 512
    constexpr float SCALING = 2.0f;     // 32/16

    const float* x      = (const float*)d_in[0];
    const float* base_w = (const float*)d_in[1];
    const float* gate_w = (const float*)d_in[2];
    const float* lora_A = (const float*)d_in[3];
    const float* lora_B = (const float*)d_in[4];
    float* out = (float*)d_out;

    char* ws = (char*)d_ws;
    __bf16* xb  = (__bf16*)ws; ws += (size_t)M * DIN * 2;       // 64 MiB
    __bf16* wb  = (__bf16*)ws; ws += (size_t)DOUT * DIN * 2;    // 32 MiB
    __bf16* lab = (__bf16*)ws; ws += (size_t)RM * DIN * 2;      //  4 MiB
    __bf16* lbb = (__bf16*)ws; ws += (size_t)DOUT * RM * 2;     //  4 MiB
    float*  wsc = (float*)ws;  ws += (size_t)M * E * 4;         //  1 MiB
    __bf16* axw = (__bf16*)ws;                                  //  8 MiB

    auto cvt = [&](const float* src, __bf16* dst, size_t n) {
        int n4 = (int)(n / 4);
        cvt_bf16_kernel<<<(n4 + 255) / 256, 256, 0, stream>>>(
            (const float4*)src, (v4bf*)dst, n4);
    };
    cvt(x,      xb,  (size_t)M * DIN);
    cvt(base_w, wb,  (size_t)DOUT * DIN);
    cvt(lora_A, lab, (size_t)RM * DIN);
    cvt(lora_B, lbb, (size_t)DOUT * RM);

    gate_kernel<<<M / 8, 256, 0, stream>>>(x, gate_w, wsc, M, DIN, SCALING);

    // axw = bf16( (xb @ lora_A^T) * wscale )   (M=8192, N=512, K=4096), fused
    wmma_gemm_kernel<<<dim3(RM / TILE_N, M / TILE_M), 256, 0, stream>>>(
        xb, lab, DIN, nullptr, nullptr, 0, (void*)axw, RM, wsc);

    // out = xb @ base_w^T + axw @ lora_B^T   (fused two-segment K loop)
    wmma_gemm_kernel<<<dim3(DOUT / TILE_N, M / TILE_M), 256, 0, stream>>>(
        xb, wb, DIN, axw, lbb, RM, (void*)out, DOUT, nullptr);
}
